// MLModel_9603546874119
// MI455X (gfx1250) — matrix operand order repaired and checked by reading, not compiled, not run on hardware
//
#include <hip/hip_runtime.h>
#include <hip/hip_bf16.h>

// ---------------------------------------------------------------------------
// kNN: dist[q][n] = ||x_q - y_n||, topk k=6 smallest per query.
// Q=64, D=3072, N=100000. GEMM via v_wmma_f32_16x16x32_bf16 with hi/lo
// bf16 split of fp32 inputs (3 WMMAs per K-chunk => ~fp32 accuracy).
// ---------------------------------------------------------------------------

typedef __attribute__((ext_vector_type(16))) __bf16 bf16x16;
typedef __attribute__((ext_vector_type(8)))  __bf16 bf16x8;
typedef __attribute__((ext_vector_type(8)))  float  f32x8;
typedef __attribute__((ext_vector_type(4)))  float  f32x4;

#define QN   64      // queries
#define DIM  3072    // feature dim (multiple of 32)
#define KC   32      // WMMA K per step (bf16)
#define NT   128     // columns per workgroup (8 waves * 16)
#define KSEL 6       // top-k

static __device__ inline f32x8 wmma_bf16(bf16x16 a, bf16x16 b, f32x8 c) {
    // 8-arg form: (neg_a, A, neg_b, B, c_mod, C, reuse_a, reuse_b)
    return __builtin_amdgcn_wmma_f32_16x16x32_bf16(
        false, a, false, b, (short)0, c, false, false);
}

// --- Kernel 1: split x into bf16 hi/lo planes + row norms -------------------
__global__ __launch_bounds__(256)
void knn_prep(const float* __restrict__ x,
              unsigned short* __restrict__ xh_u,
              unsigned short* __restrict__ xl_u,
              float* __restrict__ x2) {
    __bf16* xh = (__bf16*)xh_u;
    __bf16* xl = (__bf16*)xl_u;
    const int row = blockIdx.x;       // 0..63
    const int t   = threadIdx.x;      // 0..255
    float s = 0.f;
    for (int k = t; k < DIM; k += 256) {
        float v = x[row * DIM + k];
        __bf16 h = (__bf16)v;
        __bf16 l = (__bf16)(v - (float)h);
        xh[row * DIM + k] = h;
        xl[row * DIM + k] = l;
        s = fmaf(v, v, s);
    }
    __shared__ float red[256];
    red[t] = s;
    __syncthreads();
    for (int o = 128; o > 0; o >>= 1) {
        if (t < o) red[t] += red[t + o];
        __syncthreads();
    }
    if (t == 0) x2[row] = red[0];
}

// --- Kernel 2: dist matrix via bf16-split WMMA ------------------------------
// Each wave: 16 columns x all 64 rows (4 accumulator tiles). WG = 8 waves
// = 128 columns. y is streamed exactly once from HBM.
__global__ __launch_bounds__(256)
void knn_gemm(const unsigned short* __restrict__ xh_u,
              const unsigned short* __restrict__ xl_u,
              const float* __restrict__ x2,
              const float* __restrict__ y,
              float* __restrict__ dist, int N) {
    const __bf16* xh = (const __bf16*)xh_u;
    const __bf16* xl = (const __bf16*)xl_u;

    const int lane = threadIdx.x & 31;
    const int wave = threadIdx.x >> 5;
    const int l15  = lane & 15;
    const bool hih = lane >= 16;

    // This lane's output column (B-fragment: lanes 0-15 hold K 0..15,
    // lanes 16-31 hold K 16..31 of the same 16 columns).
    const int n    = blockIdx.x * NT + wave * 16 + l15;
    const int colc = n < N ? n : N - 1;                 // clamp for tail WG
    const float* yrow = y + (size_t)colc * DIM + (hih ? 16 : 0);

    // A-fragment K groups per lane half (contiguous: lane half h holds K 16h..16h+15).
    const int g0 = hih ? 16 : 0;
    const int g1 = hih ? 24 : 8;

    f32x8 acc[4];
    #pragma unroll
    for (int s = 0; s < 4; ++s) acc[s] = (f32x8)0.f;
    float ysq = 0.f;

    for (int k = 0; k < DIM; k += KC) {
        // ---- load 16 y floats for this lane's column / K-half ----
        union { f32x4 v4[4]; float f[16]; } bb;
        bb.v4[0] = *(const f32x4*)(yrow + k + 0);
        bb.v4[1] = *(const f32x4*)(yrow + k + 4);
        bb.v4[2] = *(const f32x4*)(yrow + k + 8);
        bb.v4[3] = *(const f32x4*)(yrow + k + 12);
        __builtin_prefetch(yrow + k + KC, 0, 0);        // global_prefetch next chunk

        bf16x16 bh, bl;
        #pragma unroll
        for (int j = 0; j < 16; ++j) {
            float f  = bb.f[j];
            __bf16 h = (__bf16)f;
            __bf16 l = (__bf16)(f - (float)h);
            bh[j] = h;
            bl[j] = l;
            ysq = fmaf(f, f, ysq);                       // ||y_n||^2 for free
        }

        // ---- 4 row-subtiles: A in pre-split bf16, 3 WMMAs each ----
        #pragma unroll
        for (int s = 0; s < 4; ++s) {
            const size_t abase = (size_t)(s * 16 + l15) * DIM + k;
            bf16x8 ah0 = *(const bf16x8*)(xh + abase + g0);
            bf16x8 ah1 = *(const bf16x8*)(xh + abase + g1);
            bf16x8 al0 = *(const bf16x8*)(xl + abase + g0);
            bf16x8 al1 = *(const bf16x8*)(xl + abase + g1);
            bf16x16 ah = __builtin_shufflevector(ah0, ah1,
                0,1,2,3,4,5,6,7,8,9,10,11,12,13,14,15);
            bf16x16 al = __builtin_shufflevector(al0, al1,
                0,1,2,3,4,5,6,7,8,9,10,11,12,13,14,15);
            acc[s] = wmma_bf16(ah, bh, acc[s]);   // hi*hi
            acc[s] = wmma_bf16(ah, bl, acc[s]);   // hi*lo
            acc[s] = wmma_bf16(al, bh, acc[s]);   // lo*hi
        }
    }

    // Combine the two K-halves of ||y||^2 (lane L and L^16 share a column).
    ysq += __shfl_xor(ysq, 16);

    // Epilogue: d2 = x2 + y2 - 2*dot ; dist = sqrt(max(d2,0))
    const int hb = hih ? 8 : 0;
    #pragma unroll
    for (int s = 0; s < 4; ++s) {
        #pragma unroll
        for (int r = 0; r < 8; ++r) {
            int row  = s * 16 + r + hb;
            float d2 = x2[row] + ysq - 2.0f * acc[s][r];
            float dv = sqrtf(fmaxf(d2, 0.0f));
            if (n < N) dist[(size_t)row * N + n] = dv;
        }
    }
}

// --- Kernel 3: per-query top-6 (one block per query) ------------------------
__global__ __launch_bounds__(256)
void knn_topk(const float* __restrict__ dist,
              float* __restrict__ outv, int* __restrict__ outi, int N) {
    const int q = blockIdx.x;
    const int t = threadIdx.x;
    float bv[KSEL];
    int   bi[KSEL];
    #pragma unroll
    for (int j = 0; j < KSEL; ++j) { bv[j] = 3.0e38f; bi[j] = 0; }

    const float* row = dist + (size_t)q * N;
    for (int nn = t; nn < N; nn += 256) {
        float v = row[nn];
        if (v < bv[KSEL - 1]) {
            float cv = v; int ci = nn;
            #pragma unroll
            for (int j = 0; j < KSEL; ++j) {
                if (cv < bv[j]) {
                    float tv = bv[j]; int ti = bi[j];
                    bv[j] = cv; bi[j] = ci; cv = tv; ci = ti;
                }
            }
        }
    }

    __shared__ float sv[256 * KSEL];
    __shared__ int   si[256 * KSEL];
    #pragma unroll
    for (int j = 0; j < KSEL; ++j) { sv[t*KSEL+j] = bv[j]; si[t*KSEL+j] = bi[j]; }
    __syncthreads();

    for (int off = 128; off > 0; off >>= 1) {
        if (t < off) {
            #pragma unroll
            for (int j = 0; j < KSEL; ++j) {
                float cv = sv[(t + off) * KSEL + j];
                int   ci = si[(t + off) * KSEL + j];
                if (cv < bv[KSEL - 1]) {
                    #pragma unroll
                    for (int m = 0; m < KSEL; ++m) {
                        if (cv < bv[m]) {
                            float tv = bv[m]; int ti = bi[m];
                            bv[m] = cv; bi[m] = ci; cv = tv; ci = ti;
                        }
                    }
                }
            }
            #pragma unroll
            for (int j = 0; j < KSEL; ++j) { sv[t*KSEL+j] = bv[j]; si[t*KSEL+j] = bi[j]; }
        }
        __syncthreads();
    }

    if (t == 0) {
        #pragma unroll
        for (int j = 0; j < KSEL; ++j) {
            outv[q * KSEL + j] = bv[j];
            outi[q * KSEL + j] = bi[j];
        }
    }
}

// ---------------------------------------------------------------------------
extern "C" void kernel_launch(void* const* d_in, const int* in_sizes, int n_in,
                              void* d_out, int out_size, void* d_ws, size_t ws_size,
                              hipStream_t stream) {
    const float* x = (const float*)d_in[0];
    const float* y = (const float*)d_in[1];
    const int N = in_sizes[1] / DIM;          // 100000

    // Workspace layout (bytes):
    //   [0)            : xh bf16  64*3072*2 = 393216
    //   [393216)       : xl bf16             393216
    //   [786432)       : x2 f32   64*4
    //   [1<<20)        : dist f32 64*N  (~25.6 MB, fits L2)
    char* ws = (char*)d_ws;
    unsigned short* xh = (unsigned short*)(ws);
    unsigned short* xl = (unsigned short*)(ws + 393216);
    float* x2   = (float*)(ws + 786432);
    float* dist = (float*)(ws + (1u << 20));

    float* outv = (float*)d_out;              // 64*6 distances
    int*   outi = (int*)((float*)d_out + QN * KSEL);  // 64*6 int32 indices

    knn_prep<<<QN, 256, 0, stream>>>(x, xh, xl, x2);
    const int nwg = (N + NT - 1) / NT;        // 782
    knn_gemm<<<nwg, 256, 0, stream>>>(xh, xl, x2, y, dist, N);
    knn_topk<<<QN, 256, 0, stream>>>(dist, outv, outi, N);
}
